// Distance_Sentences_15427522527858
// MI455X (gfx1250) — matrix operand order repaired and checked by reading, not compiled, not run on hardware
//
#include <hip/hip_runtime.h>

// ---------------------------------------------------------------------------
// Problem constants (match reference)
// ---------------------------------------------------------------------------
#define MDIM  1024          // embedding dim (GEMM K)
#define NKER  32            // distance kernel size (L1 axis)
#define OKER  32            // number of distance kernels
#define BSET  32            // batches
#define SSET  256           // set size
#define ROWS  (BSET * SSET) // 8192 GEMM rows
#define CTOT  (NKER * OKER) // 1024 GEMM cols

typedef __attribute__((ext_vector_type(16))) __bf16 v16bf;
typedef __attribute__((ext_vector_type(8)))  float  v8f;
typedef __attribute__((ext_vector_type(4)))  float  v4f;

struct U32x8 { uint4 a, b; };   // 32 bytes == one v16bf fragment

// Assemble a v16bf WMMA fragment from two 16-byte LDS chunks (pure loads,
// no per-element packing VALU).
__device__ __forceinline__ v16bf ld_frag(const unsigned short* p0,
                                         const unsigned short* p1) {
    U32x8 t;
    t.a = *(const uint4*)p0;
    t.b = *(const uint4*)p1;
    return __builtin_bit_cast(v16bf, t);
}

// Split two f32 into packed bf16 hi (truncate) and bf16 lo (residual),
// two lanes per dword: integer masks plus f32 subtract, ~6 VALU per pair.
__device__ __forceinline__ void split_pair(float f0, float f1,
                                           unsigned& hi, unsigned& lo) {
    unsigned u0 = __float_as_uint(f0);
    unsigned u1 = __float_as_uint(f1);
    hi = (u1 & 0xFFFF0000u) | (u0 >> 16);
    float l0 = f0 - __uint_as_float(u0 & 0xFFFF0000u);   // exact residual
    float l1 = f1 - __uint_as_float(u1 & 0xFFFF0000u);
    lo = (__float_as_uint(l1) & 0xFFFF0000u) | (__float_as_uint(l0) >> 16);
}

__device__ __forceinline__ void split8(const v4f& x0, const v4f& x1,
                                       uint4& hi, uint4& lo) {
    split_pair(x0.x, x0.y, hi.x, lo.x);
    split_pair(x0.z, x0.w, hi.y, lo.y);
    split_pair(x1.x, x1.y, hi.z, lo.z);
    split_pair(x1.z, x1.w, hi.w, lo.w);
}

// ---------------------------------------------------------------------------
// Kernel 1: proj[r][c] = sum_m X[r][m] * W[c][m]   (f32 in, f32 out)
// 256 threads = 8 waves; block tile 64x64; K-step 32.
// bf16 hi/lo split (done once per element, at LDS-store time) ->
// 3 x v_wmma_f32_16x16x32_bf16 per 16x16 tile per K-step.
// Fragments are read from LDS as two ds_load_b128 each (layout-native).
// ---------------------------------------------------------------------------
#define TILE  64
#define KSTEP 32
#define LPB   40   // LDS row pitch in bf16 elems: 80B = 5x16B (aligned, 0-conflict)

__global__ __launch_bounds__(256)
void gemm_bf16x3_kernel(const float* __restrict__ X,
                        const float* __restrict__ W,
                        float* __restrict__ P)
{
    __shared__ __align__(16) unsigned short Ahi[TILE * LPB];
    __shared__ __align__(16) unsigned short Alo[TILE * LPB];
    __shared__ __align__(16) unsigned short Bhi[TILE * LPB];
    __shared__ __align__(16) unsigned short Blo[TILE * LPB];

    const int row0 = blockIdx.x * TILE;
    const int col0 = blockIdx.y * TILE;
    const int t    = threadIdx.x;
    const int lane = t & 31;
    const int w    = t >> 5;
    const int half = lane >> 4;      // 0 or 1
    const int l15  = lane & 15;
    const int m0   = (w & 3) * 16;   // wave row offset in tile
    const int n0   = (w >> 2) * 32;  // wave col offset in tile

    // cooperative load mapping: thread -> (row-in-tile, 8-elem chunk of K)
    const int lrow = t >> 2;         // 0..63
    const int kq   = (t & 3) * 8;    // 0,8,16,24
    const int sst  = lrow * LPB + kq;           // LDS store offset (bf16 units)

    // Fragment offsets (bf16 units), fixed per thread:
    //  A 16x32: lane(l15,half) needs K run [16h..16h+15] (32B contiguous)
    const int aoff0 = (m0 + l15) * LPB + 16 * half;
    const int aoff1 = aoff0 + 8;
    //  B 32x16: lane(l15,half) needs K run [16h..16h+15] (32B contiguous)
    const int boff0 = (n0 + l15) * LPB + 16 * half;        // tile n0
    const int boff1 = (n0 + 16 + l15) * LPB + 16 * half;   // tile n0+16

    v8f acc0 = {};
    v8f acc1 = {};

    for (int ks = 0; ks < MDIM; ks += KSTEP) {
        // global -> regs (coalesced 128-bit loads)
        const size_t ax = (size_t)(row0 + lrow) * MDIM + ks + kq;
        const size_t bx = (size_t)(col0 + lrow) * MDIM + ks + kq;
        v4f a0 = *(const v4f*)(X + ax);
        v4f a1 = *(const v4f*)(X + ax + 4);
        v4f b0 = *(const v4f*)(W + bx);
        v4f b1 = *(const v4f*)(W + bx + 4);

        // split once per element, pack to dwords
        uint4 ahp, alp, bhp, blp;
        split8(a0, a1, ahp, alp);
        split8(b0, b1, bhp, blp);

        __syncthreads();                       // previous iter's frag reads done
        *(uint4*)(Ahi + sst) = ahp;
        *(uint4*)(Alo + sst) = alp;
        *(uint4*)(Bhi + sst) = bhp;
        *(uint4*)(Blo + sst) = blp;
        __syncthreads();                       // tiles visible

        // Fragment loads: pure ds_load_b128 pairs, already in WMMA layout.
        v16bf fah = ld_frag(Ahi + aoff0, Ahi + aoff1);
        v16bf fal = ld_frag(Alo + aoff0, Alo + aoff1);
        v16bf fbh0 = ld_frag(Bhi + boff0, Bhi + boff0 + 8);
        v16bf fbl0 = ld_frag(Blo + boff0, Blo + boff0 + 8);
        v16bf fbh1 = ld_frag(Bhi + boff1, Bhi + boff1 + 8);
        v16bf fbl1 = ld_frag(Blo + boff1, Blo + boff1 + 8);

        acc0 = __builtin_amdgcn_wmma_f32_16x16x32_bf16(false, fah, false, fbh0, (short)0, acc0, false, false);
        acc0 = __builtin_amdgcn_wmma_f32_16x16x32_bf16(false, fah, false, fbl0, (short)0, acc0, false, false);
        acc0 = __builtin_amdgcn_wmma_f32_16x16x32_bf16(false, fal, false, fbh0, (short)0, acc0, false, false);

        acc1 = __builtin_amdgcn_wmma_f32_16x16x32_bf16(false, fah, false, fbh1, (short)0, acc1, false, false);
        acc1 = __builtin_amdgcn_wmma_f32_16x16x32_bf16(false, fah, false, fbl1, (short)0, acc1, false, false);
        acc1 = __builtin_amdgcn_wmma_f32_16x16x32_bf16(false, fal, false, fbh1, (short)0, acc1, false, false);
    }

    // C/D layout: VGPR v -> row (v + 8*half), lane -> col (lane & 15)
#pragma unroll
    for (int v = 0; v < 8; ++v) {
        const size_t r = (size_t)(row0 + m0 + v + 8 * half);
        P[r * CTOT + col0 + n0 + l15]      = acc0[v];
        P[r * CTOT + col0 + n0 + 16 + l15] = acc1[v];
    }
}

// ---------------------------------------------------------------------------
// Kernel 2: out[b,i,o] = sum_k exp(-sum_n |proj[b,i,n,o] - proj[b,k,n,o]|)
// Block = (b, chunk of 8 i's), 256 threads: thread (il, o).
// Per-k 4KB tile staged in LDS, double-buffered (1 barrier/iter).
// lds[n*32 + o] reads are bank-conflict-free within a wave (o = lane).
// ---------------------------------------------------------------------------
__global__ __launch_bounds__(256)
void l1exp_kernel(const float* __restrict__ P, float* __restrict__ out)
{
    __shared__ __align__(16) float kt[2][NKER * OKER];

    const int b  = blockIdx.y;
    const int i  = blockIdx.x * 8 + (threadIdx.x >> 5);
    const int o  = threadIdx.x & 31;
    const float* Pb = P + (size_t)b * SSET * CTOT;

    // register-cache proj[b, i, :, o]
    float pi[NKER];
#pragma unroll
    for (int n = 0; n < NKER; ++n)
        pi[n] = Pb[(size_t)i * CTOT + n * OKER + o];

    // prologue: stage k = 0
    {
        v4f r0 = *(const v4f*)(Pb + (size_t)threadIdx.x * 4);
        *(v4f*)(&kt[0][threadIdx.x * 4]) = r0;
    }

    float acc = 0.0f;
    for (int k = 0; k < SSET; ++k) {
        __syncthreads();                        // kt[k&1] ready; kt[(k+1)&1] free
        v4f nxt = {};
        if (k + 1 < SSET)
            nxt = *(const v4f*)(Pb + (size_t)(k + 1) * CTOT + threadIdx.x * 4);

        const float* buf = kt[k & 1];
        float d = 0.0f;
#pragma unroll
        for (int n = 0; n < NKER; ++n)
            d += __builtin_fabsf(pi[n] - buf[n * OKER + o]);
        acc += __expf(-d);

        if (k + 1 < SSET)
            *(v4f*)(&kt[(k + 1) & 1][threadIdx.x * 4]) = nxt;
    }

    out[(size_t)b * (SSET * OKER) + (size_t)i * OKER + o] = acc;
}

// ---------------------------------------------------------------------------
// Launch
// ---------------------------------------------------------------------------
extern "C" void kernel_launch(void* const* d_in, const int* in_sizes, int n_in,
                              void* d_out, int out_size, void* d_ws, size_t ws_size,
                              hipStream_t stream)
{
    (void)in_sizes; (void)n_in; (void)out_size; (void)ws_size;

    const float* X = (const float*)d_in[0];   // (B, S, M) f32
    const float* W = (const float*)d_in[1];   // (N*O, M) f32
    float* out = (float*)d_out;               // (B, S*O) f32
    float* P   = (float*)d_ws;                // proj scratch: 8192*1024 f32 = 32 MB

    dim3 g1(ROWS / TILE, CTOT / TILE);        // 128 x 16 blocks
    gemm_bf16x3_kernel<<<g1, 256, 0, stream>>>(X, W, P);

    dim3 g2(SSET / 8, BSET);                  // 32 x 32 blocks
    l1exp_kernel<<<g2, 256, 0, stream>>>(P, out);
}
